// QuantumHybridCNN_49495203119308
// MI455X (gfx1250) — matrix operand order repaired and checked by reading, not compiled, not run on hardware
//
#include <hip/hip_runtime.h>
#include <hip/hip_bf16.h>
#include <math.h>

typedef _Float16 f16_t;
typedef __attribute__((ext_vector_type(4)))  _Float16 v4h;
typedef __attribute__((ext_vector_type(8)))  _Float16 v8h;
typedef __attribute__((ext_vector_type(16))) _Float16 v16h;
typedef __attribute__((ext_vector_type(8)))  float    v8f;

// ---------------------------------------------------------------------------
// WMMA helpers (wave32, CDNA5 ISA §7.12.2 layouts)
// Per-lane A fragment (16x32 f16) = two contiguous 16B chunks of row m=lane&15:
//   halves [8*hi, 8*hi+8) and [16+8*hi, 16+8*hi+8)   (hi = lane>>4)
// B fragments are pre-swizzled in global memory to [tile][lane][16 halves] so
// they are also two contiguous 16B loads per lane.
// ---------------------------------------------------------------------------
__device__ __forceinline__ v16h make_v16(v8h lo, v8h hi) {
    return __builtin_shufflevector(lo, hi, 0, 1, 2, 3, 4, 5, 6, 7,
                                   8, 9, 10, 11, 12, 13, 14, 15);
}

__device__ __forceinline__ v16h load_frag_2x16B(const f16_t* p0, const f16_t* p1) {
    return make_v16(*(const v8h*)p0, *(const v8h*)p1);
}

__device__ __forceinline__ v8f wmma_f16(v16h a, v16h b, v8f c) {
    return __builtin_amdgcn_wmma_f32_16x16x32_f16(
        /*neg_a=*/false, a, /*neg_b=*/false, b,
        /*c_mod=*/(short)0, c, /*reuse_a=*/false, /*reuse_b=*/false);
}

// ---------------------------------------------------------------------------
// Kernel 0: convert + swizzle weights into WMMA B-fragment-major f16 layout.
//   swizzled element (tile, lane, j) = W[kt*32 + 8*(lane>>4) + (j&7) + 16*(j>>3)][nt*16 + (lane&15)]
//   tile = kt*NTILES_N + nt ; each tile is 512 halves (32x16 KxN block).
// conv2 uses K ordering k = t*32 + ic  (t = K-chunk index, ic = index in chunk).
// ---------------------------------------------------------------------------
__global__ __launch_bounds__(256) void qh_prep_weights(
    const float* __restrict__ w2,     // (64 oc, 32 ic, 3 t)
    const float* __restrict__ fcw,    // (1024, 64)
    f16_t* __restrict__ w2sw,         // 12 tiles * 512
    f16_t* __restrict__ fcwsw) {      // 128 tiles * 512
    const int gid = blockIdx.x * blockDim.x + threadIdx.x;
    const int stride = gridDim.x * blockDim.x;

    for (int e = gid; e < 12 * 512; e += stride) {
        const int tile = e >> 9, r = e & 511;
        const int lane = r >> 4, j = r & 15;
        const int t = tile >> 2, nt = tile & 3;          // kt == tap t
        const int ic = 8 * (lane >> 4) + (j & 7) + 16 * (j >> 3); // k within chunk
        const int oc = nt * 16 + (lane & 15);
        w2sw[e] = (f16_t)w2[oc * 96 + ic * 3 + t];
    }
    for (int e = gid; e < 128 * 512; e += stride) {
        const int tile = e >> 9, r = e & 511;
        const int lane = r >> 4, j = r & 15;
        const int kt = tile >> 2, nt = tile & 3;
        const int k = kt * 32 + 8 * (lane >> 4) + (j & 7) + 16 * (j >> 3);
        const int n = nt * 16 + (lane & 15);
        fcwsw[e] = (f16_t)fcw[k * 64 + n];
    }
}

// ---------------------------------------------------------------------------
// Kernel 1: conv1+relu+pool (VALU, vector LDS stores) fused with conv2 as
//           implicit GEMM (WMMA, A-frags read straight from s_h1t with halo).
//           4 samples / block, 256 threads = 8 waves.
// ---------------------------------------------------------------------------
__global__ __launch_bounds__(256) void qh_conv_fused(
    const float* __restrict__ x,      // (B, 64)
    const float* __restrict__ w1,     // (32, 1, 3)
    const float* __restrict__ b1,     // (32,)
    const float* __restrict__ b2,     // (64,)
    const f16_t* __restrict__ w2sw,   // swizzled conv2 weights
    f16_t* __restrict__ h2) {         // (B, 1024)  flat = oc*16 + pp
    __shared__ float s_x[4][64];
    __shared__ f16_t s_h1t[4][32][32];   // [sample][pos][ic]  (pos-major!)
    __shared__ float s_w1[3][32];        // [t][oc]
    __shared__ float s_b1[32];
    __shared__ float s_b2[64];

    const int tid = threadIdx.x;
    const int b0  = blockIdx.x * 4;

    { // stage x + small weights
        const int s = tid >> 6, j = tid & 63;
        s_x[s][j] = x[(size_t)(b0 + s) * 64 + j];
        if (tid < 96) s_w1[tid % 3][tid / 3] = w1[tid];
        if (tid < 32) s_b1[tid] = b1[tid];
        if (tid < 64) s_b2[tid] = b2[tid];
    }
    __syncthreads();

    // conv1 + relu + maxpool2 -> s_h1t.  Thread = (s, p, half-of-ic):
    // reads its 4-value x window once, produces 16 contiguous ic outputs,
    // stores them as two 16B LDS writes.
    {
        const int pair = tid >> 1;            // 0..127 = (s, p)
        const int s   = pair >> 5;
        const int p   = pair & 31;
        const int oc0 = (tid & 1) * 16;
        const int j0  = 2 * p;
        const float xm1 = (j0 >= 1) ? s_x[s][j0 - 1] : 0.0f;
        const float x0  = s_x[s][j0];
        const float x1  = s_x[s][j0 + 1];
        const float x2  = (j0 + 2 < 64) ? s_x[s][j0 + 2] : 0.0f;

        v8h lo, hi;
#pragma unroll
        for (int j = 0; j < 16; ++j) {
            const int oc = oc0 + j;
            const float w0 = s_w1[0][oc], wA = s_w1[1][oc], wB = s_w1[2][oc];
            const float bb = s_b1[oc];
            const float y0 = bb + w0 * xm1 + wA * x0 + wB * x1;
            const float y1 = bb + w0 * x0  + wA * x1 + wB * x2;
            const f16_t v = (f16_t)fmaxf(fmaxf(y0, y1), 0.0f);
            if (j < 8) lo[j] = v; else hi[j - 8] = v;
        }
        v8h* dst = (v8h*)&s_h1t[s][p][oc0];
        dst[0] = lo;
        dst[1] = hi;
    }
    __syncthreads();

    // Implicit GEMM: rows = (s, p), K = t*32 + ic (3 chunks of 32), N = 64.
    // Each wave owns one 16-row M tile (half of one sample's positions).
    const int wave = tid >> 5, lane = tid & 31;
    const int hi   = lane >> 4, nlo = lane & 15;
    const int s_l  = wave >> 1;
    const int pbase = (wave & 1) * 16;
    const int p = pbase + (lane & 15);

    const v8h z8 = {f16_t(0), f16_t(0), f16_t(0), f16_t(0),
                    f16_t(0), f16_t(0), f16_t(0), f16_t(0)};
    v16h afr[3];
#pragma unroll
    for (int t = 0; t < 3; ++t) {
        const int q = p - 1 + t;                        // conv halo
        if (q >= 0 && q < 32) {
            const f16_t* row = &s_h1t[s_l][q][0];
            afr[t] = load_frag_2x16B(row + 8 * hi, row + 16 + 8 * hi);
        } else {
            afr[t] = make_v16(z8, z8);
        }
    }

    const int ppbase = (wave & 1) * 8;
#pragma unroll
    for (int nt = 0; nt < 4; ++nt) {
        v8f acc = {0.f, 0.f, 0.f, 0.f, 0.f, 0.f, 0.f, 0.f};
#pragma unroll
        for (int t = 0; t < 3; ++t) {
            const f16_t* bp = w2sw + (size_t)((t * 4 + nt) * 512) + lane * 16;
            acc = wmma_f16(afr[t], load_frag_2x16B(bp, bp + 8), acc);
        }
        // C/D layout: VGPR r <-> M = r + 8*hi. maxpool2 pairs adjacent rows:
        // acc pairs (2j,2j+1) -> pooled pp = ppbase + j + 4*hi. Packed b64 store.
        const int oc = nt * 16 + nlo;
        const float bias = s_b2[oc];
        v4h pk;
#pragma unroll
        for (int j = 0; j < 4; ++j)
            pk[j] = (f16_t)fmaxf(fmaxf(acc[2 * j], acc[2 * j + 1]) + bias, 0.0f);
        *(v4h*)(h2 + (size_t)(b0 + s_l) * 1024 + oc * 16 + ppbase + 4 * hi) = pk;
    }
}

// ---------------------------------------------------------------------------
// Kernel 2: fc GEMM (M=64,N=64,K=1024) with A-frags straight from global h2
//           and B-frags from swizzled global weights (no GEMM LDS), then
//           per-sample pre->tanh->4-qubit circuit->post->sigmoid.
// ---------------------------------------------------------------------------
__global__ __launch_bounds__(256) void qh_fc_quantum(
    const f16_t* __restrict__ h2,     // (B, 1024)
    const f16_t* __restrict__ fcwsw,  // swizzled fc weights (128 tiles * 512)
    const float* __restrict__ fcb,    // (64,)
    const float* __restrict__ prew,   // (64, 4)
    const float* __restrict__ preb,   // (4,)
    const float* __restrict__ qp,     // (24,)
    const float* __restrict__ postw,  // (4,)
    const float* __restrict__ postb,  // (1,)
    float* __restrict__ out) {        // (B,)
    __shared__ float s_out[64][65];
    __shared__ float s_fcb[64];
    __shared__ float s_prew[64][4];
    __shared__ float s_preb[4];
    __shared__ float s_qp[24];
    __shared__ float s_postw[4];
    __shared__ float s_postb;

    const int tid = threadIdx.x;
    const int b0  = blockIdx.x * 64;

    if (tid < 64)  s_fcb[tid] = fcb[tid];
    (&s_prew[0][0])[tid] = prew[tid];          // exactly 256 elements
    if (tid < 4)   s_preb[tid] = preb[tid];
    if (tid < 24)  s_qp[tid] = qp[tid];
    if (tid < 4)   s_postw[tid] = postw[tid];
    if (tid == 0)  s_postb = postb[0];

    const int wave = tid >> 5, lane = tid & 31;
    const int hi = lane >> 4, nlo = lane & 15;
    const int mt  = wave >> 1;                 // M tile 0..3
    const int ntb = (wave & 1) * 2;            // N tiles ntb, ntb+1

    const f16_t* __restrict__ arow =
        h2 + (size_t)(b0 + mt * 16 + (lane & 15)) * 1024 + 8 * hi;

    v8f acc0 = {0.f, 0.f, 0.f, 0.f, 0.f, 0.f, 0.f, 0.f};
    v8f acc1 = {0.f, 0.f, 0.f, 0.f, 0.f, 0.f, 0.f, 0.f};

#pragma unroll 4
    for (int kt = 0; kt < 32; ++kt) {
        const f16_t* ap = arow + kt * 32;
        const v16h a = load_frag_2x16B(ap, ap + 16);
        const f16_t* bp0 = fcwsw + (size_t)((kt * 4 + ntb) * 512) + lane * 16;
        const f16_t* bp1 = bp0 + 512;
        acc0 = wmma_f16(a, load_frag_2x16B(bp0, bp0 + 8), acc0);
        acc1 = wmma_f16(a, load_frag_2x16B(bp1, bp1 + 8), acc1);
    }

    __syncthreads();   // param staging visible before epilogue reads s_fcb

    // relu(acc + bias) -> s_out  (C/D layout: VGPR r <-> M = mt*16 + r + 8*hi)
    const int c0 = ntb * 16 + nlo, c1 = (ntb + 1) * 16 + nlo;
    const float bias0 = s_fcb[c0], bias1 = s_fcb[c1];
#pragma unroll
    for (int r = 0; r < 8; ++r) {
        const int m = mt * 16 + r + 8 * hi;
        s_out[m][c0] = fmaxf(acc0[r] + bias0, 0.0f);
        s_out[m][c1] = fmaxf(acc1[r] + bias1, 0.0f);
    }
    __syncthreads();

    // ---- quantum tail: one sample per lane (threads 0..63) ----
    if (tid < 64) {
        float z[4];
#pragma unroll
        for (int q = 0; q < 4; ++q) z[q] = s_preb[q];
        for (int k = 0; k < 64; ++k) {
            const float a = s_out[tid][k];
#pragma unroll
            for (int q = 0; q < 4; ++q) z[q] += a * s_prew[k][q];
        }
        float theta[4];
#pragma unroll
        for (int q = 0; q < 4; ++q)
            theta[q] = tanhf(z[q]) * 1.5707963267948966f;

        float st[16];
#pragma unroll
        for (int i = 0; i < 16; ++i) st[i] = 0.25f;

        auto ry = [&](float th, int w) {
            float sn, cs;
            __sincosf(th * 0.5f, &sn, &cs);
            const int stride = 8 >> w;
#pragma unroll
            for (int i = 0; i < 16; ++i) {
                if ((i & stride) == 0) {
                    const int j = i | stride;
                    const float a0 = st[i], a1 = st[j];
                    st[i] = cs * a0 - sn * a1;
                    st[j] = sn * a0 + cs * a1;
                }
            }
        };
        auto cnot = [&](int ctrl, int tgt) {
            const int sc = 8 >> ctrl, tg = 8 >> tgt;
#pragma unroll
            for (int i = 0; i < 16; ++i) {
                if ((i & sc) && !(i & tg)) {
                    const int j = i | tg;
                    const float tmp = st[i];
                    st[i] = st[j];
                    st[j] = tmp;
                }
            }
        };

#pragma unroll
        for (int w = 0; w < 4; ++w) ry(theta[w], w);
#pragma unroll
        for (int k = 0; k < 6; ++k) {
            cnot(0, 1);
            cnot(2, 3);
            cnot(1, 2);
#pragma unroll
            for (int w = 0; w < 4; ++w) ry(s_qp[k * 4 + w], w);
        }

        float zq[4] = {0.f, 0.f, 0.f, 0.f};
#pragma unroll
        for (int i = 0; i < 16; ++i) {
            const float p = st[i] * st[i];
#pragma unroll
            for (int q = 0; q < 4; ++q)
                zq[q] += (i & (8 >> q)) ? -p : p;
        }
        float o = s_postb;
#pragma unroll
        for (int q = 0; q < 4; ++q) o += zq[q] * s_postw[q];
        out[b0 + tid] = 1.0f / (1.0f + __expf(-o));
    }
}

// ---------------------------------------------------------------------------
// Host launcher
// ---------------------------------------------------------------------------
extern "C" void kernel_launch(void* const* d_in, const int* in_sizes, int n_in,
                              void* d_out, int out_size, void* d_ws, size_t ws_size,
                              hipStream_t stream) {
    (void)n_in; (void)out_size; (void)ws_size;
    const float* x     = (const float*)d_in[0];
    const float* w1    = (const float*)d_in[1];
    const float* b1    = (const float*)d_in[2];
    const float* w2    = (const float*)d_in[3];
    const float* b2    = (const float*)d_in[4];
    const float* fcw   = (const float*)d_in[5];
    const float* fcb   = (const float*)d_in[6];
    const float* prew  = (const float*)d_in[7];
    const float* preb  = (const float*)d_in[8];
    const float* qp    = (const float*)d_in[9];
    const float* postw = (const float*)d_in[10];
    const float* postb = (const float*)d_in[11];
    float* out = (float*)d_out;

    const int B = in_sizes[0] / 64;   // 32768

    // workspace: [w2sw 12KB][fcwsw 128KB @16KB] ... [h2 = B*1024 f16 @1MB]
    char* ws = (char*)d_ws;
    f16_t* w2sw  = (f16_t*)(ws);
    f16_t* fcwsw = (f16_t*)(ws + 16 * 1024);
    f16_t* h2    = (f16_t*)(ws + (1u << 20));

    qh_prep_weights<<<64, 256, 0, stream>>>(w2, fcw, w2sw, fcwsw);
    qh_conv_fused<<<B / 4, 256, 0, stream>>>(x, w1, b1, b2, w2sw, h2);
    qh_fc_quantum<<<B / 64, 256, 0, stream>>>(h2, fcwsw, fcb, prew, preb, qp,
                                              postw, postb, out);
}
